// MultiHeadAttention_2611340116533
// MI455X (gfx1250) — matrix operand order repaired and checked by reading, not compiled, not run on hardware
//
#include <hip/hip_runtime.h>
#include <hip/hip_bf16.h>

// MI455X / gfx1250 multi-head causal attention, bf16 WMMA pipeline.
// cvt(f32->bf16, wq pre-scaled by HS^-0.5*log2e) -> QKV GEMMs (K stored
// transposed) -> flash attention (TDM tensor_load_to_lds double-buffered
// K/V staging, WMMA scores + WMMA row-sums) -> output projection.

#define USE_TDM 1

typedef __bf16 bf16_t;
typedef __attribute__((ext_vector_type(16))) __bf16 v16bf;
typedef __attribute__((ext_vector_type(8)))  __bf16 v8bf;
typedef __attribute__((ext_vector_type(4)))  __bf16 v4bf;
typedef __attribute__((ext_vector_type(8)))  float  v8f;
typedef __attribute__((ext_vector_type(4)))  float  v4f;
typedef __attribute__((ext_vector_type(4)))  unsigned u32x4;
typedef __attribute__((ext_vector_type(8)))  unsigned u32x8;

static constexpr int BB = 2;
static constexpr int TT = 2048;
static constexpr int DD = 1024;
static constexpr int HH = 16;
static constexpr int HS = 64;

// ---------------------------------------------------------------------------
// A-matrix (16x32 bf16) fragment load, per ISA 7.12.2:
// lanes 0-15: M=lane,   elems 0..7 = K[0..7],   elems 8..15 = K[16..23]
// lanes16-31: M=lane-16,elems 0..7 = K[8..15],  elems 8..15 = K[24..31]
__device__ __forceinline__ v16bf load_a_frag(const bf16_t* rowPtr, int hi8) {
  v8bf p0 = *(const v8bf*)(rowPtr + hi8);
  v8bf p1 = *(const v8bf*)(rowPtr + 16 + hi8);
  return __builtin_shufflevector(p0, p1, 0,1,2,3,4,5,6,7, 8,9,10,11,12,13,14,15);
}

__device__ __forceinline__ v8f wmma_bf16(v16bf a, v16bf b, v8f c) {
  return __builtin_amdgcn_wmma_f32_16x16x32_bf16(false, a, false, b, (short)0, c,
                                                 false, false);
}

__device__ __forceinline__ void wait_ds_0() {
  asm volatile("s_wait_dscnt 0x0" ::: "memory");
}

// group-of-32 XOR lane swizzle (wave32): one ds_swizzle per butterfly step
template <int XORM>
__device__ __forceinline__ float swz_xor(float x) {
  constexpr int imm = (XORM << 10) | 0x1f;   // xor_mask<<10 | and_mask=0x1f
  return __int_as_float(__builtin_amdgcn_ds_swizzle(__float_as_int(x), imm));
}

#if USE_TDM
// ---------------------------------------------------------------------------
// Tensor Data Mover: DMA one 2D bf16 tile (tile_w x tile_h, row stride
// stride0 elements) from global to LDS. D# packing per ISA ch.8 bit tables.
__device__ __forceinline__ void tdm_load_2d(unsigned lds_addr, const void* gaddr,
                                            unsigned tile_w, unsigned tile_h,
                                            unsigned stride0) {
  unsigned long long ga = (unsigned long long)gaddr;
  u32x4 g0;
  g0[0] = 1u;                                   // count=1 valid user D#
  g0[1] = lds_addr;                             // LDS byte address
  g0[2] = (unsigned)ga;                         // global_addr[31:0]
  g0[3] = (unsigned)(ga >> 32) | (2u << 30);    // global_addr[56:32] | type=2
  u32x8 g1;
  g1[0] = 1u << 16;                             // data_size=1 -> 2-byte elems
  g1[1] = (tile_w & 0xFFFFu) << 16;             // tensor_dim0[15:0] (=tile_w)
  g1[2] = (tile_w >> 16) | ((tile_h & 0xFFFFu) << 16);   // dim0 hi | dim1 lo
  g1[3] = (tile_h >> 16) | (tile_w << 16);      // dim1 hi | tile_dim0
  g1[4] = tile_h;                               // tile_dim1 (tile_dim2=0)
  g1[5] = stride0;                              // tensor_dim0_stride[31:0]
  g1[6] = 0u;                                   // stride0 hi | dim1_stride lo
  g1[7] = 0u;
  asm volatile("tensor_load_to_lds %0, %1" :: "s"(g0), "s"(g1) : "memory");
}
#else
// fallback: per-lane ASYNCcnt-tracked global->LDS copies
__device__ __forceinline__ void async_load_b128(unsigned lds_off, const void* g) {
  asm volatile("global_load_async_to_lds_b128 %0, %1, off"
               :: "v"(lds_off), "v"((unsigned long long)g) : "memory");
}
__device__ __forceinline__ void wait_async_le2() {
  asm volatile("s_wait_asynccnt 0x2" ::: "memory");
}
__device__ __forceinline__ void wait_async_0() {
  asm volatile("s_wait_asynccnt 0x0" ::: "memory");
}
#endif

// online-softmax row update (base-2 domain, scores pre-scaled)
__device__ __forceinline__ void online_row(float v0, float v1, float& rmaxr,
                                           float& alphar, float& p0, float& p1) {
  float mx = fmaxf(v0, v1);
  mx = fmaxf(mx, swz_xor<1>(mx));
  mx = fmaxf(mx, swz_xor<2>(mx));
  mx = fmaxf(mx, swz_xor<4>(mx));
  mx = fmaxf(mx, swz_xor<8>(mx));
  mx = fmaxf(mx, rmaxr);
  alphar = exp2f(rmaxr - mx);
  rmaxr = mx;
  p0 = exp2f(v0 - mx);
  p1 = exp2f(v1 - mx);
}

// ---------------------------------------------------------------------------
// float32 -> bf16 conversion with optional scale, 4-wide
__global__ __launch_bounds__(256) void cvt_kernel(const float* __restrict__ in,
                                                  bf16_t* __restrict__ out,
                                                  int n4, float scale) {
  int i = blockIdx.x * blockDim.x + threadIdx.x;
  if (i < n4) {
    v4f v = *(const v4f*)(in + (size_t)i * 4);
    v4bf o;
    #pragma unroll
    for (int j = 0; j < 4; ++j) o[j] = (bf16_t)(v[j] * scale);
    *(v4bf*)(out + (size_t)i * 4) = o;
  }
}

// ---------------------------------------------------------------------------
// Generic bf16 GEMM: C[M,N] = A[M,K] * Bw[K,N], tiles 16x16, K-step 32.
template <bool OUT_BF16, bool TRANSPOSE_OUT>
__global__ __launch_bounds__(256) void gemm_kernel(
    const bf16_t* __restrict__ A, int lda, long aSZ,
    const bf16_t* __restrict__ Bw, int ldb, long bSZ,
    void* __restrict__ Cout, int ldc, long cSZ,
    const float* __restrict__ bias, int Kdim) {
  int z = blockIdx.z;
  A  += (long)z * aSZ;
  Bw += (long)z * bSZ;
  int lane = threadIdx.x & 31, wave = threadIdx.x >> 5;
  int lo = lane & 15, hi8 = (lane >> 4) * 8;
  int m0 = blockIdx.x * 128 + wave * 16;
  int n0 = blockIdx.y * 64;

  v8f acc[4] = {};
  const bf16_t* aRow = A + (long)(m0 + lo) * lda;
  for (int k0 = 0; k0 < Kdim; k0 += 32) {
    __builtin_prefetch(aRow + k0 + 64, 0, 0);  // global_prefetch_b8
    v16bf a = load_a_frag(aRow + k0, hi8);
    const bf16_t* bRow = Bw + (long)k0 * ldb + n0 + lo;  // lane <-> N column
    #pragma unroll
    for (int nc = 0; nc < 4; ++nc) {
      v16bf b;                                             // packed <-> K (table order)
      #pragma unroll
      for (int s = 0; s < 16; ++s) {
        int kk = (s & 7) + hi8 + (s >> 3) * 16;
        b[s] = bRow[(long)kk * ldb + nc * 16];
      }
      acc[nc] = wmma_bf16(a, b, acc[nc]);
    }
  }

  if (OUT_BF16) {
    bf16_t* C = (bf16_t*)Cout + (long)z * cSZ;
    #pragma unroll
    for (int nc = 0; nc < 4; ++nc) {
      int n = n0 + nc * 16 + lo;
      #pragma unroll
      for (int r = 0; r < 8; ++r) {
        int m = m0 + hi8 + r;
        long idx = TRANSPOSE_OUT ? (long)n * ldc + m : (long)m * ldc + n;
        C[idx] = (bf16_t)acc[nc][r];
      }
    }
  } else {
    float* C = (float*)Cout + (long)z * cSZ;
    #pragma unroll
    for (int nc = 0; nc < 4; ++nc) {
      int n = n0 + nc * 16 + lo;
      float bv = bias ? bias[n] : 0.f;
      #pragma unroll
      for (int r = 0; r < 8; ++r) {
        int m = m0 + hi8 + r;
        C[(long)m * ldc + n] = acc[nc][r] + bv;
      }
    }
  }
}

// ---------------------------------------------------------------------------
// Flash attention, causal, scores pre-scaled by HS^-0.5*log2e.
// grid.x = T/128, grid.z = B*H; 8 waves each owning 16 query rows.
// Key loop is uniform over the workgroup; K/V tiles staged by the TDM into
// double-buffered LDS (wave 0 issues the descriptors, TENSORcnt + barrier).
__global__ __launch_bounds__(256) void attn_kernel(
    const bf16_t* __restrict__ Q,   // [B,H,T,HS]
    const bf16_t* __restrict__ Kt,  // [B,H,HS,T]  (transposed)
    const bf16_t* __restrict__ V,   // [B,H,T,HS]
    bf16_t* __restrict__ Out) {     // [B,T,D] head-concat
  __shared__ bf16_t kbuf[2][64 * 32];   // K^T tile: 64 e-rows x 32 keys
  __shared__ bf16_t vbuf[2][32 * 64];   // V tile:   32 keys  x 64 e
  __shared__ bf16_t pbuf[8][16 * 32];   // per-wave P relayout buffer

  int tid = threadIdx.x;
  int z = blockIdx.z;
  int b = z >> 4, h = z & (HH - 1);
  const bf16_t* Qh  = Q  + (long)z * TT * HS;
  const bf16_t* Kth = Kt + (long)z * HS * TT;
  const bf16_t* Vh  = V  + (long)z * TT * HS;

  int lane = tid & 31, wave = tid >> 5;
  int lo = lane & 15, hi8 = (lane >> 4) * 8;
  int t0 = blockIdx.x * 128 + wave * 16;

  unsigned koff[2] = { (unsigned)(unsigned long long)&kbuf[0][0],
                       (unsigned)(unsigned long long)&kbuf[1][0] };
  unsigned voff[2] = { (unsigned)(unsigned long long)&vbuf[0][0],
                       (unsigned)(unsigned long long)&vbuf[1][0] };

#if USE_TDM
  auto issue_tiles = [&](int kb, int buf) {   // wave 0 only: 2 TDM descriptors
    tdm_load_2d(koff[buf], Kth + kb, /*tile_w=*/32, /*tile_h=*/64, /*stride=*/TT);
    tdm_load_2d(voff[buf], Vh + (long)kb * HS, /*tile_w=*/64, /*tile_h=*/32, HS);
  };
#else
  int ke = tid >> 2, kc = (tid & 3) * 8;
  int vr = tid >> 3, vc = (tid & 7) * 8;
  auto issue_tiles = [&](int kb, int buf) {
    async_load_b128(koff[buf] + (unsigned)(ke * 32 + kc) * 2,
                    Kth + (long)ke * TT + kb + kc);
    async_load_b128(voff[buf] + (unsigned)(vr * 64 + vc) * 2,
                    Vh + (long)(kb + vr) * HS + vc);
  };
#endif

  // Q fragments (K-dim = HS = 64 -> two A frags)
  const bf16_t* qRow = Qh + (long)(t0 + lo) * HS;
  v16bf aq0 = load_a_frag(qRow, hi8);
  v16bf aq1 = load_a_frag(qRow + 32, hi8);

  v16bf vones;
  #pragma unroll
  for (int j = 0; j < 16; ++j) vones[j] = (bf16_t)1.0f;

  v8f o0 = {}, o1 = {}, o2 = {}, o3 = {};
  float rmax[8], rsum[8];
  #pragma unroll
  for (int r = 0; r < 8; ++r) { rmax[r] = -1e30f; rsum[r] = 0.f; }

#if USE_TDM
  if (wave == 0) issue_tiles(0, 0);
#else
  issue_tiles(0, 0);
#endif
  int nblocks = (blockIdx.x + 1) * 4;      // uniform causal limit

  for (int i = 0; i < nblocks; ++i) {
    int kb = i * 32;
#if USE_TDM
    if (wave == 0) {
      if (i + 1 < nblocks) {
        issue_tiles(kb + 32, (i + 1) & 1);
        __builtin_amdgcn_s_wait_tensorcnt((short)2);  // current pair done
      } else {
        __builtin_amdgcn_s_wait_tensorcnt((short)0);
      }
    }
#else
    if (i + 1 < nblocks) { issue_tiles(kb + 32, (i + 1) & 1); wait_async_le2(); }
    else                 { wait_async_0(); }
#endif
    __syncthreads();                       // tiles visible to all waves

    if (kb <= t0 + 15) {                   // this wave has live keys here
      const bf16_t* kl = &kbuf[i & 1][0];
      const bf16_t* vl = &vbuf[i & 1][0];

      // ---- S = Q K^T (two 16x16 D-tiles over 32 keys)
      v8f s0 = {}, s1 = {};
      v16bf b00, b10, b01, b11;
      #pragma unroll
      for (int s = 0; s < 16; ++s) {
        int kk = (s & 7) + hi8 + (s >> 3) * 16;
        b00[s] = kl[kk * 32 + lo];
        b10[s] = kl[(32 + kk) * 32 + lo];
        b01[s] = kl[kk * 32 + 16 + lo];
        b11[s] = kl[(32 + kk) * 32 + 16 + lo];
      }
      s0 = wmma_bf16(aq0, b00, s0);
      s0 = wmma_bf16(aq1, b10, s0);
      s1 = wmma_bf16(aq0, b01, s1);
      s1 = wmma_bf16(aq1, b11, s1);

      float alpha[8];
      bf16_t* myp = pbuf[wave];
      if (kb + 31 > t0) {                  // causal boundary blocks only
        int nA0 = kb + lo, nA1 = kb + 16 + lo;
        #pragma unroll
        for (int r = 0; r < 8; ++r) {
          int m = t0 + hi8 + r;
          float v0 = (nA0 <= m) ? s0[r] : -1e30f;
          float v1 = (nA1 <= m) ? s1[r] : -1e30f;
          float p0, p1;
          online_row(v0, v1, rmax[r], alpha[r], p0, p1);
          myp[(hi8 + r) * 32 + lo]      = (bf16_t)p0;
          myp[(hi8 + r) * 32 + 16 + lo] = (bf16_t)p1;
          o0[r] *= alpha[r]; o1[r] *= alpha[r];
          o2[r] *= alpha[r]; o3[r] *= alpha[r];
        }
      } else {                             // steady state: no mask math
        #pragma unroll
        for (int r = 0; r < 8; ++r) {
          float p0, p1;
          online_row(s0[r], s1[r], rmax[r], alpha[r], p0, p1);
          myp[(hi8 + r) * 32 + lo]      = (bf16_t)p0;
          myp[(hi8 + r) * 32 + 16 + lo] = (bf16_t)p1;
          o0[r] *= alpha[r]; o1[r] *= alpha[r];
          o2[r] *= alpha[r]; o3[r] *= alpha[r];
        }
      }
      wait_ds_0();                         // same-wave LDS RAW drain

      // ---- relayout P to A-frag; row sums via WMMA against ones matrix
      v16bf ap = load_a_frag(myp + lo * 32, hi8);
      v8f srow = {};
      srow = wmma_bf16(ap, vones, srow);   // every lane gets its 8 row sums
      v16bf bv0, bv1, bv2, bv3;
      #pragma unroll
      for (int s = 0; s < 16; ++s) {
        int kk = (s & 7) + hi8 + (s >> 3) * 16;
        bv0[s] = vl[kk * 64 + 0 + lo];
        bv1[s] = vl[kk * 64 + 16 + lo];
        bv2[s] = vl[kk * 64 + 32 + lo];
        bv3[s] = vl[kk * 64 + 48 + lo];
      }
      o0 = wmma_bf16(ap, bv0, o0);
      o1 = wmma_bf16(ap, bv1, o1);
      o2 = wmma_bf16(ap, bv2, o2);
      o3 = wmma_bf16(ap, bv3, o3);
      #pragma unroll
      for (int r = 0; r < 8; ++r) rsum[r] = rsum[r] * alpha[r] + srow[r];
    }
    __syncthreads();                       // done reading before next overwrite
  }

  // ---- normalize + head-concat store: Out[b, t, h*HS + e]
  bf16_t* outBase = Out + (long)b * TT * DD + (long)h * HS;
  #pragma unroll
  for (int r = 0; r < 8; ++r) {
    float inv = 1.0f / rsum[r];
    bf16_t* orow = outBase + (long)(t0 + hi8 + r) * DD;
    orow[lo]      = (bf16_t)(o0[r] * inv);
    orow[16 + lo] = (bf16_t)(o1[r] * inv);
    orow[32 + lo] = (bf16_t)(o2[r] * inv);
    orow[48 + lo] = (bf16_t)(o3[r] * inv);
  }
}

// ---------------------------------------------------------------------------
extern "C" void kernel_launch(void* const* d_in, const int* in_sizes, int n_in,
                              void* d_out, int out_size, void* d_ws, size_t ws_size,
                              hipStream_t stream) {
  (void)in_sizes; (void)n_in; (void)out_size; (void)ws_size;
  const float* x  = (const float*)d_in[0];
  const float* wq = (const float*)d_in[1];
  const float* wk = (const float*)d_in[2];
  const float* wv = (const float*)d_in[3];
  const float* wp = (const float*)d_in[4];
  const float* bp = (const float*)d_in[5];

  char* ws = (char*)d_ws;
  auto take = [&](size_t elems) { bf16_t* p = (bf16_t*)ws; ws += elems * sizeof(bf16_t); return p; };
  bf16_t* xb  = take((size_t)BB * TT * DD);
  bf16_t* wqb = take((size_t)HH * DD * HS);
  bf16_t* wkb = take((size_t)HH * DD * HS);
  bf16_t* wvb = take((size_t)HH * DD * HS);
  bf16_t* wpb = take((size_t)DD * DD);
  bf16_t* qb  = take((size_t)BB * HH * TT * HS);
  bf16_t* ktb = take((size_t)BB * HH * HS * TT);
  bf16_t* vb  = take((size_t)BB * HH * TT * HS);
  bf16_t* aob = take((size_t)BB * TT * DD);

  auto cvt = [&](const float* in, bf16_t* out, int n, float s) {
    int n4 = n / 4;
    cvt_kernel<<<(n4 + 255) / 256, 256, 0, stream>>>(in, out, n4, s);
  };
  const float qscale = 0.125f * 1.44269504088896340736f;  // HS^-0.5 * log2(e)
  cvt(x, xb, BB * TT * DD, 1.0f);
  cvt(wq, wqb, HH * DD * HS, qscale);   // scores arrive softmax-ready
  cvt(wk, wkb, HH * DD * HS, 1.0f);
  cvt(wv, wvb, HH * DD * HS, 1.0f);
  cvt(wp, wpb, DD * DD, 1.0f);

  for (int b = 0; b < BB; ++b) {
    dim3 g(TT / 128, 1, HH);
    const bf16_t* Ab = xb + (long)b * TT * DD;
    gemm_kernel<true, false><<<g, 256, 0, stream>>>(
        Ab, DD, 0L, wqb, HS, (long)DD * HS,
        qb + (long)b * HH * TT * HS, HS, (long)TT * HS, nullptr, DD);
    gemm_kernel<true, true><<<g, 256, 0, stream>>>(   // K stored transposed
        Ab, DD, 0L, wkb, HS, (long)DD * HS,
        ktb + (long)b * HH * HS * TT, TT, (long)HS * TT, nullptr, DD);
    gemm_kernel<true, false><<<g, 256, 0, stream>>>(
        Ab, DD, 0L, wvb, HS, (long)DD * HS,
        vb + (long)b * HH * TT * HS, HS, (long)TT * HS, nullptr, DD);
  }

  {
    dim3 g(TT / 128, 1, BB * HH);
    attn_kernel<<<g, 256, 0, stream>>>(qb, ktb, vb, aob);
  }

  {
    dim3 g((BB * TT) / 128, DD / 64, 1);
    gemm_kernel<false, false><<<g, 256, 0, stream>>>(
        aob, DD, 0L, wpb, DD, 0L, d_out, DD, 0L, bp, DD);
  }
}
